// AttentionHPROJ_9337258901903
// MI455X (gfx1250) — matrix operand order repaired and checked by reading, not compiled, not run on hardware
//
#include <hip/hip_runtime.h>
#include <hip/hip_bf16.h>

// ---------------------------------------------------------------------------
// CDNA5 (gfx1250) causal attention + head-wise projection, bf16 WMMA pipeline.
// B=2, T=2048, C=1024, h=16, hd=64.
// Two-phase (explicitly unrolled) double-buffered async global->LDS staging.
// ---------------------------------------------------------------------------

typedef __attribute__((ext_vector_type(16))) __bf16       v16bf;
typedef __attribute__((ext_vector_type(8)))  float        v8f;
typedef __attribute__((ext_vector_type(8)))  unsigned int v8u;
typedef __attribute__((ext_vector_type(4)))  int          v4i;
typedef unsigned short u16;

#define BQ   2
#define TT   2048
#define CC   1024
#define NH   16
#define HD   64

// ---- float -> bf16 conversion: prefer v_cvt_pk_bf16_f32 -------------------
#if defined(__gfx1250__) && __has_builtin(__builtin_amdgcn_cvt_pk_bf16_f32)
__device__ inline unsigned int pack2(float a, float b) {
  auto r = __builtin_amdgcn_cvt_pk_bf16_f32(a, b);
  unsigned int u;
  __builtin_memcpy(&u, &r, 4);
  return u;
}
__device__ inline u16 f2bf(float f) { return (u16)(pack2(f, f) & 0xffffu); }
#else
__device__ inline u16 f2bf(float f) {
  unsigned int u = __float_as_uint(f);
  u = u + 0x7fffu + ((u >> 16) & 1u);
  return (u16)(u >> 16);
}
__device__ inline unsigned int pack2(float a, float b) {
  return (unsigned int)f2bf(a) | ((unsigned int)f2bf(b) << 16);
}
#endif

// ---- async global->LDS staging (gfx1250 ASYNCcnt path) --------------------
#if defined(__gfx1250__) && __has_builtin(__builtin_amdgcn_global_load_async_to_lds_b128)
#define USE_ASYNC 1
typedef __attribute__((address_space(1))) v4i* as1_v4i;
typedef __attribute__((address_space(3))) v4i* as3_v4i;
__device__ inline void async_b128(const u16* g, u16* l) {
  __builtin_amdgcn_global_load_async_to_lds_b128(
      (as1_v4i)(v4i*)(void*)const_cast<u16*>(g),
      (as3_v4i)(v4i*)(void*)l, 0, 0);
}
#if __has_builtin(__builtin_amdgcn_s_wait_asynccnt)
#define WAIT_ASYNC() __builtin_amdgcn_s_wait_asynccnt(0)
#else
#define WAIT_ASYNC() asm volatile("s_wait_asynccnt 0x0" ::: "memory")
#endif
#else
#define USE_ASYNC 0
#define WAIT_ASYNC() do {} while (0)
#endif

// A-matrix fragment, 16x32 bf16, from row-major [16][ld] (ld multiple of 8).
// Lanes 0-15: row M=lane, K={0..7,16..23}; lanes 16-31: K={8..15,24..31}.
__device__ inline v8u load_frag_a(const u16* base, int ld, int lane) {
  const u16* p = base + (lane & 15) * ld + ((lane & 16) ? 8 : 0);
  uint4 lo = *(const uint4*)(p);
  uint4 hi = *(const uint4*)(p + 16);
  v8u r;
  r[0] = lo.x; r[1] = lo.y; r[2] = lo.z; r[3] = lo.w;
  r[4] = hi.x; r[5] = hi.y; r[6] = hi.z; r[7] = hi.w;
  return r;
}

// B-matrix fragment, 32x16 bf16, from [n][k] row-major source.
// Lanes 0-15: column N=lane, K={0..7,16..23}; lanes 16-31: K={8..15,24..31}.
__device__ inline v8u load_frag_b(const u16* base, int ld, int lane) {
  const u16* p = base + (lane & 15) * ld + ((lane & 16) ? 8 : 0);
  uint4 lo = *(const uint4*)(p);
  uint4 hi = *(const uint4*)(p + 16);
  v8u r;
  r[0] = lo.x; r[1] = lo.y; r[2] = lo.z; r[3] = lo.w;
  r[4] = hi.x; r[5] = hi.y; r[6] = hi.z; r[7] = hi.w;
  return r;
}

__device__ inline v8f wmma_bf16(v8u a, v8u b, v8f c) {
  return __builtin_amdgcn_wmma_f32_16x16x32_bf16(
      false, __builtin_bit_cast(v16bf, a),
      false, __builtin_bit_cast(v16bf, b),
      (short)0, c, false, false);
}

// One K=32 step of the 32x64 wave tile: 2x4 WMMAs from LDS tiles.
__device__ inline void gemm_step(const u16* Atile, const u16* Btile,
                                 int wm, int wn, int lane, v8f (&acc)[2][4]) {
  v8u a0 = load_frag_a(&Atile[(wm * 32 +  0) * 32], 32, lane);
  v8u a1 = load_frag_a(&Atile[(wm * 32 + 16) * 32], 32, lane);
#pragma unroll
  for (int j = 0; j < 4; ++j) {
    v8u b = load_frag_b(&Btile[(wn * 64 + j * 16) * 32], 32, lane);
    acc[0][j] = wmma_bf16(a0, b, acc[0][j]);
    acc[1][j] = wmma_bf16(a1, b, acc[1][j]);
  }
}

// ---------------------------------------------------------------------------
// Kernel 0: bulk fp32 -> bf16 convert (8 elements / thread, b128 in/out).
// ---------------------------------------------------------------------------
__global__ __launch_bounds__(256) void cvt_f32_bf16(const float* __restrict__ src,
                                                    u16* __restrict__ dst) {
  size_t i = ((size_t)blockIdx.x * 256 + threadIdx.x) * 8;
  float4 f0 = *(const float4*)(src + i);
  float4 f1 = *(const float4*)(src + i + 4);
  uint4 o;
  o.x = pack2(f0.x, f0.y); o.y = pack2(f0.z, f0.w);
  o.z = pack2(f1.x, f1.y); o.w = pack2(f1.z, f1.w);
  *(uint4*)(dst + i) = o;
}

// ---------------------------------------------------------------------------
// Kernel 1: repack W_proj (fp32 [C][C]) into Wp2 bf16 [d][k], k = h*64+v:
//   Wp2[d][h*64+v] = W_proj[h*64 + d/16][(d%16)*64 + v]
// ---------------------------------------------------------------------------
__global__ __launch_bounds__(256) void wp2_repack(const float* __restrict__ Wp,
                                                  u16* __restrict__ Wp2) {
  int idx = blockIdx.x * 256 + threadIdx.x;   // 1M elements
  int d = idx >> 10;
  int k = idx & 1023;
  int hh = k >> 6, vv = k & 63;
  float v = Wp[(size_t)(hh * 64 + (d >> 4)) * CC + (d & 15) * 64 + vv];
  Wp2[idx] = f2bf(v);
}

// ---------------------------------------------------------------------------
// GEMM staging: copy a 128x32 bf16 tile [row][k] into LDS (async if possible).
// src points at element (row0, kt) of a [rows][1024] matrix.
// ---------------------------------------------------------------------------
__device__ inline void stage_tile(const u16* __restrict__ src, u16* tile, int tid) {
#pragma unroll
  for (int i = 0; i < 2; ++i) {
    int seg = tid + i * 256;          // 512 segments of 8 bf16
    int r  = seg >> 2;
    int c8 = (seg & 3) * 8;
    const u16* g = src + (size_t)r * CC + c8;
#if USE_ASYNC
    async_b128(g, &tile[r * 32 + c8]);
#else
    *(uint4*)&tile[r * 32 + c8] = *(const uint4*)g;
#endif
  }
}

// ---------------------------------------------------------------------------
// Kernel 2: QKV GEMM (bf16).  qkv[m][n] = sum_c xb[m][c] * Wab[n][c]
// M=4096, N=3072, K=1024.  Block tile 128x128, K-tile 32, two-phase
// double-buffered async staging, 8 waves (4x2), wave tile 32x64.
// Output scattered to q [b,h,t,v], k [b,h,t,v] (b16), vT [b,h,v,t] (b128).
// ---------------------------------------------------------------------------
__global__ __launch_bounds__(256) void qkv_gemm(const u16* __restrict__ xb,
                                                const u16* __restrict__ Wab,
                                                u16* __restrict__ q_buf,
                                                u16* __restrict__ k_buf,
                                                u16* __restrict__ vT_buf) {
  __shared__ u16 At[2][128 * 32];   // [m][k], double buffered
  __shared__ u16 Bt[2][128 * 32];   // [n][k]
  const int tid  = threadIdx.x;
  const int lane = tid & 31;
  const int wave = tid >> 5;
  const int wm = wave & 3;
  const int wn = wave >> 2;
  const int row0 = blockIdx.y * 128;
  const int n0   = blockIdx.x * 128;
  const u16* Asrc = xb  + (size_t)row0 * CC;
  const u16* Bsrc = Wab + (size_t)n0   * CC;

  v8f acc[2][4] = {};

  // prologue: stage first K-tile into buffer 0
  stage_tile(Asrc, At[0], tid);
  stage_tile(Bsrc, Bt[0], tid);
  WAIT_ASYNC();
  __syncthreads();

  for (int kt = 0; kt < CC; kt += 64) {
    // phase A: compute buf0, async-stage kt+32 into buf1
    stage_tile(Asrc + kt + 32, At[1], tid);
    stage_tile(Bsrc + kt + 32, Bt[1], tid);
    if (kt + 96 < CC) {  // feed L2 ahead
      __builtin_prefetch(Asrc + (size_t)(tid >> 1) * CC + kt + 96, 0, 1);
      __builtin_prefetch(Bsrc + (size_t)(tid >> 1) * CC + kt + 96, 0, 1);
    }
    gemm_step(At[0], Bt[0], wm, wn, lane, acc);
    WAIT_ASYNC();
    __syncthreads();

    // phase B: compute buf1, async-stage kt+64 into buf0
    if (kt + 64 < CC) {
      stage_tile(Asrc + kt + 64, At[0], tid);
      stage_tile(Bsrc + kt + 64, Bt[0], tid);
    }
    gemm_step(At[1], Bt[1], wm, wn, lane, acc);
    WAIT_ASYNC();
    __syncthreads();
  }

  // Scatter D (row = r + 8*(lane>=16), col = lane&15) into q/k/vT layouts.
  const int m_base = row0 + wm * 32;
  const int n_base = n0 + wn * 64;
#pragma unroll
  for (int tm = 0; tm < 2; ++tm) {
#pragma unroll
    for (int j = 0; j < 4; ++j) {
      int n  = n_base + j * 16 + (lane & 15);
      int wh = n >> 10, rem = n & 1023;       // uniform per fragment
      int hh = rem >> 6, vv = rem & 63;
      int m0 = m_base + tm * 16 + ((lane & 16) ? 8 : 0);
      int b  = m0 >> 11, t0 = m0 & 2047;
      size_t bh = (size_t)(b * NH + hh);
      if (wh == 2) {
        // vT: 8 consecutive t at fixed v -> one b128 store
        uint4 pk;
        pk.x = pack2(acc[tm][j][0], acc[tm][j][1]);
        pk.y = pack2(acc[tm][j][2], acc[tm][j][3]);
        pk.z = pack2(acc[tm][j][4], acc[tm][j][5]);
        pk.w = pack2(acc[tm][j][6], acc[tm][j][7]);
        *(uint4*)&vT_buf[(bh * HD + vv) * TT + t0] = pk;
      } else {
        u16* dst = (wh == 0) ? q_buf : k_buf;
#pragma unroll
        for (int r = 0; r < 8; ++r) {
          dst[(bh * TT + (t0 + r)) * HD + vv] = f2bf(acc[tm][j][r]);
        }
      }
    }
  }
}

// ---------------------------------------------------------------------------
// Kernel 3: causal flash attention (no 1/sqrt(hd) scaling, per reference).
// 8 waves/block; each wave owns 16 query rows of one (b,h).  S and O via
// bf16 WMMA (fp32 accum), online softmax in fp32, P routed through a
// per-wave LDS tile (D-layout -> A-layout).  O emitted bf16 [b][t][h*64+v].
// ---------------------------------------------------------------------------
__global__ __launch_bounds__(256) void attn(const u16* __restrict__ q_buf,
                                            const u16* __restrict__ k_buf,
                                            const u16* __restrict__ vT_buf,
                                            u16* __restrict__ O_buf) {
  __shared__ u16 P[8][16 * 72];          // per-wave P tile, ld=72
  const int tid  = threadIdx.x;
  const int lane = tid & 31;
  const int wave = tid >> 5;
  const int bh   = blockIdx.x >> 4;      // b*16 + h
  const int qblk = blockIdx.x & 15;
  const int b = bh >> 4, h = bh & 15;
  const int qw = qblk * 128 + wave * 16;

  const u16* qp = q_buf + ((size_t)bh * TT + qw) * HD;
  v8u qa0 = load_frag_a(qp,      HD, lane);
  v8u qa1 = load_frag_a(qp + 32, HD, lane);

  v8f o[4] = {};
  float rmax[8], rsum[8];
#pragma unroll
  for (int r = 0; r < 8; ++r) { rmax[r] = -__builtin_inff(); rsum[r] = 0.0f; }

  u16* Pw = &P[wave][0];
  const int jmax = ((qw + 15) >> 6) << 6;

  for (int j0 = 0; j0 <= jmax; j0 += 64) {
    const u16* kp = k_buf + ((size_t)bh * TT + j0) * HD;
    v8f s[4];
#pragma unroll
    for (int nf = 0; nf < 4; ++nf) {
      v8u b0 = load_frag_b(kp + (size_t)(nf * 16) * HD,      HD, lane);
      v8u b1 = load_frag_b(kp + (size_t)(nf * 16) * HD + 32, HD, lane);
      v8f z = {};
      z = wmma_bf16(qa0, b0, z);
      s[nf] = wmma_bf16(qa1, b1, z);
    }

    if (j0 + 63 > qw) {            // causal mask only on diagonal block
#pragma unroll
      for (int nf = 0; nf < 4; ++nf) {
        int jg = j0 + nf * 16 + (lane & 15);
#pragma unroll
        for (int r = 0; r < 8; ++r) {
          int qg = qw + r + ((lane & 16) ? 8 : 0);
          if (jg > qg) s[nf][r] = -__builtin_inff();
        }
      }
    }

    // online softmax: rows live in 16-lane halves (row = r + 8*(lane>=16))
#pragma unroll
    for (int r = 0; r < 8; ++r) {
      float m = fmaxf(fmaxf(s[0][r], s[1][r]), fmaxf(s[2][r], s[3][r]));
      m = fmaxf(m, __shfl_xor(m, 1, 32));
      m = fmaxf(m, __shfl_xor(m, 2, 32));
      m = fmaxf(m, __shfl_xor(m, 4, 32));
      m = fmaxf(m, __shfl_xor(m, 8, 32));
      float newm = fmaxf(rmax[r], m);
      float corr = __expf(rmax[r] - newm);
      rmax[r] = newm;
      float psum = 0.0f;
#pragma unroll
      for (int nf = 0; nf < 4; ++nf) {
        float p = __expf(s[nf][r] - newm);
        s[nf][r] = p;
        psum += p;
      }
      psum += __shfl_xor(psum, 1, 32);
      psum += __shfl_xor(psum, 2, 32);
      psum += __shfl_xor(psum, 4, 32);
      psum += __shfl_xor(psum, 8, 32);
      rsum[r] = rsum[r] * corr + psum;
#pragma unroll
      for (int nf = 0; nf < 4; ++nf) o[nf][r] *= corr;
    }

    // D-layout -> A-layout via per-wave LDS tile (bf16)
#pragma unroll
    for (int nf = 0; nf < 4; ++nf) {
#pragma unroll
      for (int r = 0; r < 8; ++r) {
        int m = r + ((lane & 16) ? 8 : 0);
        Pw[m * 72 + nf * 16 + (lane & 15)] = f2bf(s[nf][r]);
      }
    }
    v8u pa0 = load_frag_a(Pw,      72, lane);
    v8u pa1 = load_frag_a(Pw + 32, 72, lane);

    const u16* vp = vT_buf + (size_t)bh * HD * TT + j0;   // [v][t], ld=T
#pragma unroll
    for (int nf = 0; nf < 4; ++nf) {
      v8u vb0 = load_frag_b(vp + (size_t)(nf * 16) * TT,      TT, lane);
      v8u vb1 = load_frag_b(vp + (size_t)(nf * 16) * TT + 32, TT, lane);
      o[nf] = wmma_bf16(pa0, vb0, o[nf]);
      o[nf] = wmma_bf16(pa1, vb1, o[nf]);
    }
  }

  // normalize and emit O as bf16 [b][t][h*64+v]
#pragma unroll
  for (int r = 0; r < 8; ++r) {
    float inv = 1.0f / rsum[r];
    int t = qw + r + ((lane & 16) ? 8 : 0);
    size_t row = (size_t)b * TT + t;
#pragma unroll
    for (int nf = 0; nf < 4; ++nf) {
      int col = h * 64 + nf * 16 + (lane & 15);
      O_buf[row * CC + col] = f2bf(o[nf][r] * inv);
    }
  }
}

// ---------------------------------------------------------------------------
// Kernel 4: output projection. Y[m][d] = sum_k O[m][k] * Wp2[d][k]
// M=4096, N=1024, K=1024; bf16 inputs, fp32 output; two-phase staging.
// ---------------------------------------------------------------------------
__global__ __launch_bounds__(256) void proj_gemm(const u16* __restrict__ A,
                                                 const u16* __restrict__ Bm,
                                                 float* __restrict__ out) {
  __shared__ u16 At[2][128 * 32];
  __shared__ u16 Bt[2][128 * 32];
  const int tid  = threadIdx.x;
  const int lane = tid & 31;
  const int wave = tid >> 5;
  const int wm = wave & 3;
  const int wn = wave >> 2;
  const int row0 = blockIdx.y * 128;
  const int n0   = blockIdx.x * 128;
  const u16* Asrc = A  + (size_t)row0 * CC;
  const u16* Bsrc = Bm + (size_t)n0   * CC;

  v8f acc[2][4] = {};

  stage_tile(Asrc, At[0], tid);
  stage_tile(Bsrc, Bt[0], tid);
  WAIT_ASYNC();
  __syncthreads();

  for (int kt = 0; kt < CC; kt += 64) {
    stage_tile(Asrc + kt + 32, At[1], tid);
    stage_tile(Bsrc + kt + 32, Bt[1], tid);
    if (kt + 96 < CC) {
      __builtin_prefetch(Asrc + (size_t)(tid >> 1) * CC + kt + 96, 0, 1);
      __builtin_prefetch(Bsrc + (size_t)(tid >> 1) * CC + kt + 96, 0, 1);
    }
    gemm_step(At[0], Bt[0], wm, wn, lane, acc);
    WAIT_ASYNC();
    __syncthreads();

    if (kt + 64 < CC) {
      stage_tile(Asrc + kt + 64, At[0], tid);
      stage_tile(Bsrc + kt + 64, Bt[0], tid);
    }
    gemm_step(At[1], Bt[1], wm, wn, lane, acc);
    WAIT_ASYNC();
    __syncthreads();
  }

  const int m_base = row0 + wm * 32;
  const int n_base = n0 + wn * 64;
#pragma unroll
  for (int tm = 0; tm < 2; ++tm) {
#pragma unroll
    for (int j = 0; j < 4; ++j) {
#pragma unroll
      for (int r = 0; r < 8; ++r) {
        int m = m_base + tm * 16 + r + ((lane & 16) ? 8 : 0);
        int n = n_base + j * 16 + (lane & 15);
        out[(size_t)m * CC + n] = acc[tm][j][r];
      }
    }
  }
}

// ---------------------------------------------------------------------------
// Launch.  Workspace (bf16): q 4M | k 4M | vT 4M | O 4M | Wp2 1M |
//                            x_bf 4M | Wa_bf 3M  = 24M u16 = 48 MB.
// ---------------------------------------------------------------------------
extern "C" void kernel_launch(void* const* d_in, const int* in_sizes, int n_in,
                              void* d_out, int out_size, void* d_ws, size_t ws_size,
                              hipStream_t stream) {
  (void)in_sizes; (void)n_in; (void)out_size; (void)ws_size;
  const float* x      = (const float*)d_in[0];
  const float* W_attn = (const float*)d_in[1];
  const float* W_proj = (const float*)d_in[2];
  float* out = (float*)d_out;

  u16* ws     = (u16*)d_ws;
  u16* q_buf  = ws;                             // [B,h,T,hd]
  u16* k_buf  = ws + (size_t)4  * 1024 * 1024;  // [B,h,T,hd]
  u16* vT_buf = ws + (size_t)8  * 1024 * 1024;  // [B,h,hd,T]
  u16* O_buf  = ws + (size_t)12 * 1024 * 1024;  // [B*T, C]
  u16* Wp2    = ws + (size_t)16 * 1024 * 1024;  // [C, C] ([d][k])
  u16* x_bf   = ws + (size_t)17 * 1024 * 1024;  // [B*T, C]
  u16* Wa_bf  = ws + (size_t)21 * 1024 * 1024;  // [3C, C]

  cvt_f32_bf16<<<2048, 256, 0, stream>>>(x, x_bf);         // 4M elems
  cvt_f32_bf16<<<1536, 256, 0, stream>>>(W_attn, Wa_bf);   // 3M elems
  wp2_repack<<<4096, 256, 0, stream>>>(W_proj, Wp2);
  qkv_gemm<<<dim3(24, 32), 256, 0, stream>>>(x_bf, Wa_bf, q_buf, k_buf, vT_buf);
  attn<<<512, 256, 0, stream>>>(q_buf, k_buf, vT_buf, O_buf);
  proj_gemm<<<dim3(8, 32), 256, 0, stream>>>(O_buf, Wp2, out);
}
